// DecoderLayer_20478404067850
// MI455X (gfx1250) — matrix operand order repaired and checked by reading, not compiled, not run on hardware
//
#include <hip/hip_runtime.h>

typedef _Float16 f16;
typedef __attribute__((ext_vector_type(16))) _Float16 v16h;
typedef __attribute__((ext_vector_type(8)))  _Float16 v8h;
typedef __attribute__((ext_vector_type(4)))  _Float16 v4h;
typedef __attribute__((ext_vector_type(8)))  float    v8f;

constexpr int D_MODEL = 768;
constexpr int NHEADS  = 12;
constexpr int DEPTH   = 64;
constexpr int DFF     = 3072;
constexpr int BATCH   = 8;
constexpr int SEQ     = 1024;
constexpr int NTOK    = BATCH * SEQ;           // 8192
constexpr float EPS   = 1e-6f;

// ---------------------------------------------------------------------------
// Build a 16-half fragment from two 16-byte LDS chunks (keeps 16B alignment).
__device__ __forceinline__ v16h make_frag(const f16* lo, const f16* hi) {
  v8h l = *(const v8h*)lo;
  v8h h = *(const v8h*)hi;
  v16h r;
#pragma unroll
  for (int i = 0; i < 8; ++i) { r[i] = l[i]; r[i + 8] = h[i]; }
  return r;
}

__device__ __forceinline__ v8f wmma16x16x32(v16h a, v16h b, v8f c) {
  // D = A*B + C, f16 inputs, f32 accumulate
  return __builtin_amdgcn_wmma_f32_16x16x32_f16(false, a, false, b, (short)0, c,
                                                false, false);
}

// 16B async copy global -> LDS (ASYNCcnt-tracked, bypasses VGPRs).
// LDS offset = low 32 bits of the generic pointer (flat->LDS truncation rule).
__device__ __forceinline__ void async_ld_b128(f16* lds, const f16* gptr) {
  unsigned lo = (unsigned)(uintptr_t)lds;
  asm volatile("global_load_async_to_lds_b128 %0, %1, off"
               :: "v"(lo), "v"(gptr) : "memory");
}
__device__ __forceinline__ void wait_async0() {
  asm volatile("s_wait_asynccnt 0" ::: "memory");
}

// ---------------------------------------------------------------------------
// f32 -> f16 cast (vectorized x4)
__global__ void k_cast(const float* __restrict__ in, f16* __restrict__ out, int n4) {
  int i = blockIdx.x * blockDim.x + threadIdx.x;
  if (i < n4) {
    float4 v = ((const float4*)in)[i];
    v4h o;
    o[0] = (f16)v.x; o[1] = (f16)v.y; o[2] = (f16)v.z; o[3] = (f16)v.w;
    ((v4h*)out)[i] = o;
  }
}

// ---------------------------------------------------------------------------
// Generic WMMA GEMM:  C[M,N] = A[M,K] (f16, row major) * B[K,N] (f16, row major)
//                              + bias[N] (f32), optional ReLU.
// Block tile 64(M) x 128(N), 256 threads = 8 waves, wave tile 16x64.
// Double-buffered LDS; A tile staged with async-to-LDS, B tile transposed
// through VGPRs (needed for the contiguous-K B-fragment layout).
template <bool RELU>
__global__ __launch_bounds__(256)
void k_gemm(const f16* __restrict__ A, const f16* __restrict__ B,
            const float* __restrict__ bias,
            float* __restrict__ Cf, f16* __restrict__ Ch,
            int M, int N, int K)
{
  __shared__ __align__(16) f16 As[2][64][40];    // 32 K-halves + pad
  __shared__ __align__(16) f16 Bt[2][128][40];   // transposed: [n][k], + pad

  const int tid  = threadIdx.x;
  const int lane = tid & 31;
  const int wave = tid >> 5;
  const int wm   = wave & 3;        // 4 M-slices of 16
  const int wn   = wave >> 2;       // 2 N-slices of 64
  const int m0   = blockIdx.y * 64;
  const int n0   = blockIdx.x * 128;

  const int r16 = lane & 15;
  const int hiL = lane >> 4;

  v8f zero = {};
  v8f acc[4];
#pragma unroll
  for (int i = 0; i < 4; ++i) acc[i] = zero;

  const int arow = tid >> 2;            // 0..63
  const int ak   = (tid & 3) * 8;       // 0,8,16,24
  const int bk0  = tid >> 4;            // 0..15
  const int bn   = (tid & 15) * 8;      // 0..120

  auto stageA = [&](int buf, int k0) {
    async_ld_b128(&As[buf][arow][ak], &A[(size_t)(m0 + arow) * K + k0 + ak]);
  };
  auto stageB = [&](int buf, int k0) {
#pragma unroll
    for (int it = 0; it < 2; ++it) {
      int kk = bk0 + it * 16;
      v8h bv = *(const v8h*)&B[(size_t)(k0 + kk) * N + n0 + bn];
#pragma unroll
      for (int i = 0; i < 8; ++i) Bt[buf][bn + i][kk] = bv[i];
    }
  };

  stageA(0, 0);
  stageB(0, 0);
  wait_async0();
  __syncthreads();

  int buf = 0;
  for (int k0 = 0; k0 < K; k0 += 32) {
    const int nk = k0 + 32;
    if (nk < K) {                     // prefetch next tiles into the other buffer
      stageA(buf ^ 1, nk);
      stageB(buf ^ 1, nk);
    }

    // A fragment: lanes 0-15 -> K {0..15}, lanes 16-31 -> K {16..31}
    const f16* ap = &As[buf][wm * 16 + r16][hiL * 16];
    v16h a = make_frag(ap, ap + 8);
#pragma unroll
    for (int nf = 0; nf < 4; ++nf) {
      // B fragment: 16 contiguous K for fixed n, split at lane 16
      const f16* bp = &Bt[buf][wn * 64 + nf * 16 + r16][hiL * 16];
      v16h b = make_frag(bp, bp + 8);
      acc[nf] = wmma16x16x32(a, b, acc[nf]);
    }

    wait_async0();                    // next-tile async copy landed
    __syncthreads();                  // everyone done reading buf / writing buf^1
    buf ^= 1;
  }

  // epilogue: C layout = lane n (lane&15), VGPR r -> row r + 8*(lane>>4)
#pragma unroll
  for (int nf = 0; nf < 4; ++nf) {
    int col = n0 + wn * 64 + nf * 16 + r16;
    float bval = bias ? bias[col] : 0.f;
#pragma unroll
    for (int r = 0; r < 8; ++r) {
      int row = m0 + wm * 16 + r + 8 * hiL;
      float v = acc[nf][r] + bval;
      if (RELU) v = fmaxf(v, 0.f);
      size_t idx = (size_t)row * N + col;
      if (Cf) Cf[idx] = v;
      if (Ch) Ch[idx] = (f16)v;
    }
  }
}

// ---------------------------------------------------------------------------
// Fused causal flash attention. Q/K/V are [B,S,H*64] f16 (head-major cols).
// Block = 128 threads (4 waves); block handles one (b,h,64 q-rows) tile.
// kv tiles of 32 keys, online softmax, O accumulated in f32.
// KV tiles double-buffered; K staged via async-to-LDS, V transposed manually.
__global__ __launch_bounds__(128)
void k_attn(const f16* __restrict__ Q, const f16* __restrict__ Kg,
            const f16* __restrict__ Vg, f16* __restrict__ ctx)
{
  __shared__ __align__(16) f16 Qs[64][72];        // [q][d]
  __shared__ __align__(16) f16 Ks[2][32][72];     // [k][d]  (row-major == B layout)
  __shared__ __align__(16) f16 Vt[2][64][40];     // [d][k]  (transposed V)
  __shared__ __align__(16) f16 Ps[4][16][40];     // wave-private P tiles [q][k]

  const int tid  = threadIdx.x;
  const int lane = tid & 31;
  const int w    = tid >> 5;
  const int r16  = lane & 15;
  const int hiL  = lane >> 4;

  const int q0 = blockIdx.x * 64;
  const int h  = blockIdx.y;
  const int b  = blockIdx.z;
  const size_t base = (size_t)b * SEQ * D_MODEL + (size_t)h * DEPTH;

  // stage Q tile (64 x 64 halves) via async copy
#pragma unroll
  for (int c = 0; c < 4; ++c) {
    int idx = tid + c * 128;
    int row = idx >> 3, ch = (idx & 7) * 8;
    async_ld_b128(&Qs[row][ch], &Q[base + (size_t)(q0 + row) * D_MODEL + ch]);
  }

  auto stageKV = [&](int buf, int kv0) {
#pragma unroll
    for (int c = 0; c < 2; ++c) {
      int idx = tid + c * 128;
      int row = idx >> 3, ch = (idx & 7) * 8;
      async_ld_b128(&Ks[buf][row][ch],
                    &Kg[base + (size_t)(kv0 + row) * D_MODEL + ch]);
      v8h vv = *(const v8h*)&Vg[base + (size_t)(kv0 + row) * D_MODEL + ch];
#pragma unroll
      for (int i = 0; i < 8; ++i) Vt[buf][ch + i][row] = vv[i];
    }
  };

  float mrow[8], lrow[8];
  v8f zero = {};
  v8f oacc[4];
#pragma unroll
  for (int r = 0; r < 8; ++r) { mrow[r] = -1e30f; lrow[r] = 0.f; }
#pragma unroll
  for (int d = 0; d < 4; ++d) oacc[d] = zero;

  const float scale = 0.125f;   // 1/sqrt(64)
  const int   kvend = q0 + 64;

  stageKV(0, 0);
  wait_async0();                // Q + first KV tile landed
  __syncthreads();

  int buf = 0;
  for (int kv0 = 0; kv0 < kvend; kv0 += 32) {
    if (kv0 + 32 < kvend) stageKV(buf ^ 1, kv0 + 32);

    // S(16q x 32k) = Q K^T : two k=32 depth steps, two key sub-tiles of 16
    v8f s[2]; s[0] = zero; s[1] = zero;
#pragma unroll
    for (int ds = 0; ds < 2; ++ds) {
      const f16* ap = &Qs[w * 16 + r16][ds * 32 + hiL * 16];
      v16h a = make_frag(ap, ap + 8);
#pragma unroll
      for (int nf = 0; nf < 2; ++nf) {
        const f16* bp = &Ks[buf][nf * 16 + r16][ds * 32 + hiL * 16];
        v16h bf = make_frag(bp, bp + 8);
        s[nf] = wmma16x16x32(a, bf, s[nf]);
      }
    }

    // scale + causal mask (reference adds -1e9 to masked logits)
    float sv[2][8];
#pragma unroll
    for (int nf = 0; nf < 2; ++nf) {
      int kk = kv0 + nf * 16 + r16;
#pragma unroll
      for (int r = 0; r < 8; ++r) {
        int qg = q0 + w * 16 + r + 8 * hiL;
        sv[nf][r] = s[nf][r] * scale + (kk > qg ? -1e9f : 0.f);
      }
    }

    // online softmax: rows live across 16 lanes of a half-wave
#pragma unroll
    for (int r = 0; r < 8; ++r) {
      float mt = fmaxf(sv[0][r], sv[1][r]);
#pragma unroll
      for (int o = 1; o < 16; o <<= 1) mt = fmaxf(mt, __shfl_xor(mt, o, 16));
      float mn = fmaxf(mrow[r], mt);
      float sf = __expf(mrow[r] - mn);
      mrow[r] = mn;
      float p0 = __expf(sv[0][r] - mn);
      float p1 = __expf(sv[1][r] - mn);
      float rs = p0 + p1;
#pragma unroll
      for (int o = 1; o < 16; o <<= 1) rs += __shfl_xor(rs, o, 16);
      lrow[r] = lrow[r] * sf + rs;
      int prow = r + 8 * hiL;
      Ps[w][prow][r16]      = (f16)p0;   // wave-private: DS ops are in-order
      Ps[w][prow][16 + r16] = (f16)p1;
#pragma unroll
      for (int d = 0; d < 4; ++d) oacc[d][r] *= sf;
    }

    // O += P(16x32) * V(32x64)
    const f16* pp = &Ps[w][r16][hiL * 16];
    v16h pa = make_frag(pp, pp + 8);
#pragma unroll
    for (int d = 0; d < 4; ++d) {
      const f16* bp = &Vt[buf][d * 16 + r16][hiL * 16];
      v16h bf = make_frag(bp, bp + 8);
      oacc[d] = wmma16x16x32(pa, bf, oacc[d]);
    }

    wait_async0();
    __syncthreads();
    buf ^= 1;
  }

  // normalize and write context as f16 (feeds the Wo GEMM)
#pragma unroll
  for (int d = 0; d < 4; ++d) {
#pragma unroll
    for (int r = 0; r < 8; ++r) {
      int row = q0 + w * 16 + r + 8 * hiL;
      int col = h * DEPTH + d * 16 + r16;
      ctx[((size_t)b * SEQ + row) * D_MODEL + col] = (f16)(oacc[d][r] / lrow[r]);
    }
  }
}

// ---------------------------------------------------------------------------
// out = LayerNorm(X + Y) * gamma + beta ; one block (256 thr) per 768-row.
__global__ __launch_bounds__(256)
void k_add_ln(const float* __restrict__ X, const float* __restrict__ Y,
              const float* __restrict__ gamma, const float* __restrict__ beta,
              float* __restrict__ Of, f16* __restrict__ Oh)
{
  __shared__ float red[8];
  __shared__ float stat[2];
  const int tid = threadIdx.x, lane = tid & 31, w = tid >> 5;
  const size_t base = (size_t)blockIdx.x * D_MODEL;

  float v[3];
#pragma unroll
  for (int i = 0; i < 3; ++i) {
    int c = tid + i * 256;
    v[i] = X[base + c] + Y[base + c];
  }
  float s = v[0] + v[1] + v[2];
#pragma unroll
  for (int o = 16; o >= 1; o >>= 1) s += __shfl_xor(s, o, 32);
  if (lane == 0) red[w] = s;
  __syncthreads();
  if (tid == 0) {
    float t = 0;
#pragma unroll
    for (int i = 0; i < 8; ++i) t += red[i];
    stat[0] = t * (1.f / D_MODEL);
  }
  __syncthreads();
  const float mean = stat[0];

  float q = 0;
#pragma unroll
  for (int i = 0; i < 3; ++i) { float d = v[i] - mean; q += d * d; }
#pragma unroll
  for (int o = 16; o >= 1; o >>= 1) q += __shfl_xor(q, o, 32);
  __syncthreads();
  if (lane == 0) red[w] = q;
  __syncthreads();
  if (tid == 0) {
    float t = 0;
#pragma unroll
    for (int i = 0; i < 8; ++i) t += red[i];
    stat[1] = rsqrtf(t * (1.f / D_MODEL) + EPS);
  }
  __syncthreads();
  const float inv = stat[1];
#pragma unroll
  for (int i = 0; i < 3; ++i) {
    int c = tid + i * 256;
    float o = gamma[c] * (v[i] - mean) * inv + beta[c];
    Of[base + c] = o;
    if (Oh) Oh[base + c] = (f16)o;
  }
}

// ---------------------------------------------------------------------------
extern "C" void kernel_launch(void* const* d_in, const int* in_sizes, int n_in,
                              void* d_out, int out_size, void* d_ws, size_t ws_size,
                              hipStream_t stream)
{
  (void)in_sizes; (void)n_in; (void)out_size; (void)ws_size;
  const float* x   = (const float*)d_in[0];
  // d_in[1] = look_ahead_mask (causal, recomputed in-kernel)
  const float* wq  = (const float*)d_in[2];
  const float* bq  = (const float*)d_in[3];
  const float* wk  = (const float*)d_in[4];
  const float* bk  = (const float*)d_in[5];
  const float* wv  = (const float*)d_in[6];
  const float* bv  = (const float*)d_in[7];
  const float* wo  = (const float*)d_in[8];
  const float* bo  = (const float*)d_in[9];
  const float* w1  = (const float*)d_in[10];
  const float* b1  = (const float*)d_in[11];
  const float* w2  = (const float*)d_in[12];
  const float* b2  = (const float*)d_in[13];
  const float* g1  = (const float*)d_in[14];
  const float* be1 = (const float*)d_in[15];
  const float* g2  = (const float*)d_in[16];
  const float* be2 = (const float*)d_in[17];
  float* out = (float*)d_out;

  char* ws = (char*)d_ws;
  size_t off = 0;
  auto alloc = [&](size_t bytes) -> void* {
    void* p = ws + off;
    off += (bytes + 255) & ~(size_t)255;
    return p;
  };

  const size_t ND  = (size_t)NTOK * D_MODEL;
  f16*   xh   = (f16*)alloc(ND * 2);                       // also reused as ctx
  f16*   wqh  = (f16*)alloc((size_t)D_MODEL * D_MODEL * 2);
  f16*   wkh  = (f16*)alloc((size_t)D_MODEL * D_MODEL * 2);
  f16*   wvh  = (f16*)alloc((size_t)D_MODEL * D_MODEL * 2);
  f16*   woh  = (f16*)alloc((size_t)D_MODEL * D_MODEL * 2);
  f16*   w1h  = (f16*)alloc((size_t)D_MODEL * DFF * 2);
  f16*   w2h  = (f16*)alloc((size_t)DFF * D_MODEL * 2);
  f16*   qh   = (f16*)alloc(ND * 2);                       // reused as out1_h
  f16*   kh   = (f16*)alloc(ND * 2);
  f16*   vh   = (f16*)alloc(ND * 2);
  float* attn = (float*)alloc(ND * 4);                     // reused as ffn
  float* out1 = (float*)alloc(ND * 4);
  f16*   h1h  = (f16*)alloc((size_t)NTOK * DFF * 2);

  f16*   ctxh  = xh;     // xh dead after V projection
  f16*   out1h = qh;     // qh dead after attention
  float* ffn   = attn;   // attn dead after first LayerNorm

  auto cast = [&](const float* src, f16* dst, size_t n) {
    int n4 = (int)(n / 4);
    k_cast<<<(n4 + 255) / 256, 256, 0, stream>>>(src, dst, n4);
  };
  cast(x,  xh,  ND);
  cast(wq, wqh, (size_t)D_MODEL * D_MODEL);
  cast(wk, wkh, (size_t)D_MODEL * D_MODEL);
  cast(wv, wvh, (size_t)D_MODEL * D_MODEL);
  cast(wo, woh, (size_t)D_MODEL * D_MODEL);
  cast(w1, w1h, (size_t)D_MODEL * DFF);
  cast(w2, w2h, (size_t)DFF * D_MODEL);

  const dim3 gD(D_MODEL / 128, NTOK / 64);   // N=768 GEMMs
  const dim3 gF(DFF / 128,     NTOK / 64);   // N=3072 GEMM

  // Q,K,V projections (f16 outputs)
  k_gemm<false><<<gD, 256, 0, stream>>>(xh, wqh, bq, nullptr, qh, NTOK, D_MODEL, D_MODEL);
  k_gemm<false><<<gD, 256, 0, stream>>>(xh, wkh, bk, nullptr, kh, NTOK, D_MODEL, D_MODEL);
  k_gemm<false><<<gD, 256, 0, stream>>>(xh, wvh, bv, nullptr, vh, NTOK, D_MODEL, D_MODEL);

  // fused causal attention -> ctx (f16)
  k_attn<<<dim3(SEQ / 64, NHEADS, BATCH), 128, 0, stream>>>(qh, kh, vh, ctxh);

  // output projection (f32 out for residual)
  k_gemm<false><<<gD, 256, 0, stream>>>(ctxh, woh, bo, attn, nullptr, NTOK, D_MODEL, D_MODEL);

  // out1 = LN(x + attn)
  k_add_ln<<<NTOK, 256, 0, stream>>>(x, attn, g1, be1, out1, out1h);

  // FFN
  k_gemm<true ><<<gF, 256, 0, stream>>>(out1h, w1h, b1, nullptr, h1h, NTOK, DFF, D_MODEL);
  k_gemm<false><<<gD, 256, 0, stream>>>(h1h, w2h, b2, ffn, nullptr, NTOK, D_MODEL, DFF);

  // out2 = LN(out1 + ffn)
  k_add_ln<<<NTOK, 256, 0, stream>>>(out1, ffn, g2, be2, out, nullptr);
}
